// DeeperGCN_50964081935488
// MI455X (gfx1250) — matrix operand order repaired and checked by reading, not compiled, not run on hardware
//
#include <hip/hip_runtime.h>
#include <hip/hip_bf16.h>

// ---------------- problem constants ----------------
constexpr int NN = 20000;
constexpr int EE = 320000;
constexpr int FF = 14;
constexpr int DD = 128;
constexpr int LL = 28;
constexpr int HH = 256;   // 2*D

typedef __bf16 v16bf __attribute__((ext_vector_type(16)));
typedef float  v8f   __attribute__((ext_vector_type(8)));
typedef unsigned short ushort_t;

// ---------------- helpers ----------------
__device__ inline ushort_t f2bf(float f) {
    unsigned u = __float_as_uint(f);
    unsigned r = (u + 0x7FFFu + ((u >> 16) & 1u)) >> 16;
    return (ushort_t)r;
}
__device__ inline __bf16 bfbits(ushort_t s) {
    return __builtin_bit_cast(__bf16, s);
}

// signed/unsigned split trick: works for mixed-sign float max with -inf init
__device__ inline void atomicMaxF(float* addr, float val) {
    if (val >= 0.0f) atomicMax((int*)addr, __float_as_int(val));
    else             atomicMin((unsigned int*)addr, __float_as_uint(val));
}

// A fragment: 16x32 bf16 from LDS, row m, per CDNA5 layout (lane half g selects K-halves).
// Within a lane the two 8-element runs are contiguous in K -> compiler emits ds_load_b128 x2.
__device__ inline v16bf loadA(const ushort_t* base, int m, int ld, int k0, int g) {
    v16bf A;
#pragma unroll
    for (int e = 0; e < 16; ++e) {
        int v = e >> 1, p = e & 1;
        int kl = (v < 4) ? (g * 8 + 2 * v + p) : (16 + g * 8 + 2 * (v - 4) + p);
        A[e] = bfbits(base[m * ld + k0 + kl]);
    }
    return A;
}

// B fragment from fragment-packed weights: tile-linear, 32 lanes x 16 bf16 (32B per lane)
__device__ inline v16bf loadBpk(const ushort_t* base, int tile, int lane) {
    return ((const v16bf*)base)[tile * 32 + lane];
}

// ---------------- kernels ----------------

// Repack fp32 weight matrix [nLayers, K, Ncols] into WMMA B-fragment order (bf16):
// dst[((l*nKT+kt)*nCT+ct)*32 + lane][e] = bf16(src[l][kt*32 + (e<8 ? g*8 + e : 16 + g*8 + (e-8))][ct*16 + (lane&15)])
__global__ void k_pack_b(const float* __restrict__ src, ushort_t* __restrict__ dst,
                         int K, int Ncols, int nLayers) {
    int gid = blockIdx.x * blockDim.x + threadIdx.x;
    int nKT = K >> 5, nCT = Ncols >> 4;
    int total = nLayers * nKT * nCT * 32;
    if (gid >= total) return;
    int lane = gid & 31;
    int tile = gid >> 5;                 // (l*nKT + kt)*nCT + ct
    int ct = tile % nCT;
    int rest = tile / nCT;
    int kt = rest % nKT;
    int l = rest / nKT;
    int g = lane >> 4, n = lane & 15;
    const float* s = src + (size_t)l * K * Ncols;
    ushort_t* d = dst + (size_t)gid * 16;
#pragma unroll
    for (int e = 0; e < 16; ++e) {
        int v = e >> 1, p = e & 1;
        int k = kt * 32 + ((v < 4) ? (g * 8 + 2 * v + p) : (16 + g * 8 + 2 * (v - 4) + p));
        d[e] = f2bf(s[k * Ncols + ct * 16 + n]);
    }
}

// node encoder: h = x @ node_w + node_b   (N x 14 @ 14 x 128)
__global__ void k_node_enc(const float* __restrict__ x, const float* __restrict__ w,
                           const float* __restrict__ b, float* __restrict__ h) {
    int gid = blockIdx.x * blockDim.x + threadIdx.x;
    if (gid >= NN * DD) return;
    int n = gid >> 7, d = gid & 127;
    float acc = b[d];
#pragma unroll
    for (int f = 0; f < FF; ++f) acc += x[n * FF + f] * w[f * DD + d];
    h[gid] = acc;
}

// z = relu(LN(h)*g+b); one wave (32 lanes) per node, 8 nodes per 256-thread block
__global__ __launch_bounds__(256) void k_prenorm(const float* __restrict__ h,
                                                 const float* __restrict__ g,
                                                 const float* __restrict__ b,
                                                 float* __restrict__ z) {
    int w = threadIdx.x >> 5, lane = threadIdx.x & 31;
    int n = blockIdx.x * 8 + w;
    if (n >= NN) return;
    float v[4], s = 0.f, sq = 0.f;
#pragma unroll
    for (int j = 0; j < 4; ++j) {
        v[j] = h[n * DD + lane + 32 * j];
        s += v[j]; sq += v[j] * v[j];
    }
#pragma unroll
    for (int off = 16; off > 0; off >>= 1) {
        s  += __shfl_xor(s, off, 32);
        sq += __shfl_xor(sq, off, 32);
    }
    float mean = s * (1.0f / DD);
    float var  = sq * (1.0f / DD) - mean * mean;
    float rstd = rsqrtf(var + 1e-5f);
#pragma unroll
    for (int j = 0; j < 4; ++j) {
        int d = lane + 32 * j;
        float o = (v[j] - mean) * rstd * g[d] + b[d];
        z[n * DD + d] = fmaxf(o, 0.0f);
    }
}

// reset segment buffers
__global__ void k_init_seg(float* __restrict__ smax, float* __restrict__ sumex,
                           float* __restrict__ wsum) {
    int i = blockIdx.x * blockDim.x + threadIdx.x;
    if (i >= NN * DD) return;
    smax[i]  = -__builtin_inff();
    sumex[i] = 0.0f;
    wsum[i]  = 0.0f;
}

// pass 1: msg = relu(z[src]+ea)+eps, s = msg*t, segment max (32 threads/edge, 4 dims each)
__global__ void k_edge_max(const float* __restrict__ z, const float* __restrict__ eattr,
                           const int* __restrict__ ei, const float* __restrict__ ew,
                           const float* __restrict__ eb, const float* __restrict__ t, int l,
                           float* __restrict__ smax) {
    int gid = blockIdx.x * blockDim.x + threadIdx.x;
    int e = gid >> 5;
    if (e >= EE) return;
    int d0 = (gid & 31) * 4;
    int s = ei[e], dn = ei[EE + e];
    float tl = t[l];
    float a0 = eattr[e * 4 + 0], a1 = eattr[e * 4 + 1];
    float a2 = eattr[e * 4 + 2], a3 = eattr[e * 4 + 3];
#pragma unroll
    for (int k = 0; k < 4; ++k) {
        int d = d0 + k;
        float ea = a0 * ew[0 * DD + d] + a1 * ew[1 * DD + d] +
                   a2 * ew[2 * DD + d] + a3 * ew[3 * DD + d] + eb[d];
        float m = fmaxf(z[s * DD + d] + ea, 0.0f) + 1e-7f;
        atomicMaxF(&smax[dn * DD + d], m * tl);
    }
}

// pass 2: ex = exp(s - smax[dst]); sum ex and sum msg*ex
__global__ void k_edge_expsum(const float* __restrict__ z, const float* __restrict__ eattr,
                              const int* __restrict__ ei, const float* __restrict__ ew,
                              const float* __restrict__ eb, const float* __restrict__ t, int l,
                              const float* __restrict__ smax, float* __restrict__ sumex,
                              float* __restrict__ wsum) {
    int gid = blockIdx.x * blockDim.x + threadIdx.x;
    int e = gid >> 5;
    if (e >= EE) return;
    int d0 = (gid & 31) * 4;
    int s = ei[e], dn = ei[EE + e];
    float tl = t[l];
    float a0 = eattr[e * 4 + 0], a1 = eattr[e * 4 + 1];
    float a2 = eattr[e * 4 + 2], a3 = eattr[e * 4 + 3];
#pragma unroll
    for (int k = 0; k < 4; ++k) {
        int d = d0 + k;
        float ea = a0 * ew[0 * DD + d] + a1 * ew[1 * DD + d] +
                   a2 * ew[2 * DD + d] + a3 * ew[3 * DD + d] + eb[d];
        float m = fmaxf(z[s * DD + d] + ea, 0.0f) + 1e-7f;
        float ex = __expf(m * tl - smax[dn * DD + d]);
        atomicAdd(&sumex[dn * DD + d], ex);
        atomicAdd(&wsum[dn * DD + d], m * ex);
    }
}

// fused: out = wsum/(sumex+1e-16)+z; h1 = out@W1+b1; LN+relu; h2 = a@W2+b2; h (+)= h2
// block = 256 threads (8 waves) handles a 16-node tile; WMMA bf16 16x16x32.
// Weights are fragment-packed: W1pk has 4x16 tiles/layer, W2pk has 8x8 tiles/layer.
__global__ __launch_bounds__(256) void k_mlp(const float* __restrict__ z,
                                             const float* __restrict__ wsum,
                                             const float* __restrict__ sumex,
                                             const ushort_t* __restrict__ W1pk,
                                             const float* __restrict__ b1,
                                             const float* __restrict__ g1,
                                             const float* __restrict__ be1,
                                             const ushort_t* __restrict__ W2pk,
                                             const float* __restrict__ b2,
                                             float* __restrict__ h, int residual) {
    __shared__ ushort_t sA[16 * DD];   // out tile, bf16
    __shared__ float    sH[16 * HH];   // h1, f32
    __shared__ ushort_t sB[16 * HH];   // relu(LN(h1)), bf16
    int tid = threadIdx.x;
    int n0 = blockIdx.x * 16;

    // stage A: finalize aggregation + root add, convert to bf16
#pragma unroll
    for (int i = 0; i < 8; ++i) {
        int idx = tid + i * 256;
        int r = idx >> 7, d = idx & 127;
        int n = n0 + r;
        float val = wsum[n * DD + d] / (sumex[n * DD + d] + 1e-16f) + z[n * DD + d];
        sA[r * DD + d] = f2bf(val);
    }
    __syncthreads();

    int w = tid >> 5, lane = tid & 31;
    int m16 = lane & 15, g = lane >> 4;

    // GEMM1: 16x256, wave handles col tiles w and w+8 (16 tiles of 16 cols)
#pragma unroll
    for (int tt = 0; tt < 2; ++tt) {
        int ct = w + tt * 8;
        v8f acc = {};
#pragma unroll
        for (int kk = 0; kk < DD; kk += 32) {
            int kt = kk >> 5;
            if (kt < 3)  // prefetch next k-step's fragment (global_prefetch_b8)
                __builtin_prefetch(&W1pk[(size_t)(((kt + 1) * 16 + ct) * 32 + lane) * 16], 0, 3);
            v16bf A = loadA(sA, m16, DD, kk, g);
            v16bf B = loadBpk(W1pk, kt * 16 + ct, lane);
            acc = __builtin_amdgcn_wmma_f32_16x16x32_bf16(false, A, false, B,
                                                          (short)0, acc, false, false);
        }
#pragma unroll
        for (int r = 0; r < 8; ++r) {
            int mm = 8 * g + r, col = ct * 16 + m16;
            sH[mm * HH + col] = acc[r] + b1[col];
        }
    }
    __syncthreads();

    // LayerNorm + ReLU over H=256 per row; wave handles rows 2w, 2w+1
#pragma unroll
    for (int rr = 0; rr < 2; ++rr) {
        int row = 2 * w + rr;
        float vv[8], s = 0.f, sq = 0.f;
#pragma unroll
        for (int j = 0; j < 8; ++j) {
            vv[j] = sH[row * HH + lane + 32 * j];
            s += vv[j]; sq += vv[j] * vv[j];
        }
#pragma unroll
        for (int off = 16; off > 0; off >>= 1) {
            s  += __shfl_xor(s, off, 32);
            sq += __shfl_xor(sq, off, 32);
        }
        float mean = s * (1.0f / HH);
        float var  = sq * (1.0f / HH) - mean * mean;
        float rstd = rsqrtf(var + 1e-5f);
#pragma unroll
        for (int j = 0; j < 8; ++j) {
            int col = lane + 32 * j;
            float o = (vv[j] - mean) * rstd * g1[col] + be1[col];
            sB[row * HH + col] = f2bf(fmaxf(o, 0.0f));
        }
    }
    __syncthreads();

    // GEMM2: 16x128, wave w computes col tile w, K=256 (8 k-steps)
    {
        v8f acc = {};
#pragma unroll
        for (int kk = 0; kk < HH; kk += 32) {
            int kt = kk >> 5;
            if (kt < 7)
                __builtin_prefetch(&W2pk[(size_t)(((kt + 1) * 8 + w) * 32 + lane) * 16], 0, 3);
            v16bf A = loadA(sB, m16, HH, kk, g);
            v16bf B = loadBpk(W2pk, kt * 8 + w, lane);
            acc = __builtin_amdgcn_wmma_f32_16x16x32_bf16(false, A, false, B,
                                                          (short)0, acc, false, false);
        }
#pragma unroll
        for (int r = 0; r < 8; ++r) {
            int mm = 8 * g + r, col = w * 16 + m16;
            int n = n0 + mm;
            float val = acc[r] + b2[col];
            if (residual) h[n * DD + col] += val;
            else          h[n * DD + col]  = val;
        }
    }
}

// final: out = z @ lin_w + lin_b (N x 128 @ 128 x 128), WMMA bf16, packed weights (4x8 tiles)
__global__ __launch_bounds__(256) void k_final(const float* __restrict__ z,
                                               const ushort_t* __restrict__ lwpk,
                                               const float* __restrict__ lb,
                                               float* __restrict__ out) {
    __shared__ ushort_t sA[16 * DD];
    int tid = threadIdx.x;
    int n0 = blockIdx.x * 16;
#pragma unroll
    for (int i = 0; i < 8; ++i) {
        int idx = tid + i * 256;
        int r = idx >> 7, d = idx & 127;
        sA[r * DD + d] = f2bf(z[(n0 + r) * DD + d]);
    }
    __syncthreads();
    int w = tid >> 5, lane = tid & 31;
    int m16 = lane & 15, g = lane >> 4;
    v8f acc = {};
#pragma unroll
    for (int kk = 0; kk < DD; kk += 32) {
        int kt = kk >> 5;
        v16bf A = loadA(sA, m16, DD, kk, g);
        v16bf B = loadBpk(lwpk, kt * 8 + w, lane);
        acc = __builtin_amdgcn_wmma_f32_16x16x32_bf16(false, A, false, B,
                                                      (short)0, acc, false, false);
    }
#pragma unroll
    for (int r = 0; r < 8; ++r) {
        int mm = 8 * g + r, col = w * 16 + m16;
        out[(n0 + mm) * DD + col] = acc[r] + lb[col];
    }
}

// ---------------- host launcher ----------------
extern "C" void kernel_launch(void* const* d_in, const int* in_sizes, int n_in,
                              void* d_out, int out_size, void* d_ws, size_t ws_size,
                              hipStream_t stream) {
    (void)in_sizes; (void)n_in; (void)out_size; (void)ws_size;
    const float* x      = (const float*)d_in[0];
    const float* eattr  = (const float*)d_in[1];
    const int*   ei     = (const int*)d_in[2];
    const float* node_w = (const float*)d_in[3];
    const float* node_b = (const float*)d_in[4];
    const float* edge_w = (const float*)d_in[5];
    const float* edge_b = (const float*)d_in[6];
    const float* t      = (const float*)d_in[7];
    const float* W1     = (const float*)d_in[8];
    const float* b1     = (const float*)d_in[9];
    const float* g1     = (const float*)d_in[10];
    const float* be1    = (const float*)d_in[11];
    const float* W2     = (const float*)d_in[12];
    const float* b2     = (const float*)d_in[13];
    const float* norm_g = (const float*)d_in[14];
    const float* norm_b = (const float*)d_in[15];
    const float* lin_w  = (const float*)d_in[16];
    const float* lin_b  = (const float*)d_in[17];
    float* out = (float*)d_out;

    char* ws = (char*)d_ws;
    size_t off = 0;
    auto alloc = [&](size_t bytes) -> void* {
        void* p = ws + off;
        off = (off + bytes + 255) & ~(size_t)255;
        return p;
    };
    float*    h     = (float*)alloc((size_t)NN * DD * 4);
    float*    zb    = (float*)alloc((size_t)NN * DD * 4);
    float*    smax  = (float*)alloc((size_t)NN * DD * 4);
    float*    sumex = (float*)alloc((size_t)NN * DD * 4);
    float*    wsum  = (float*)alloc((size_t)NN * DD * 4);
    ushort_t* W1pk  = (ushort_t*)alloc((size_t)LL * DD * HH * 2);   // 4x16 tiles/layer
    ushort_t* W2pk  = (ushort_t*)alloc((size_t)LL * HH * DD * 2);   // 8x8 tiles/layer
    ushort_t* lwpk  = (ushort_t*)alloc((size_t)DD * DD * 2);        // 4x8 tiles

    // weight repack into WMMA fragment order (bf16, L2-resident afterwards)
    {
        int t1 = LL * (DD / 32) * (HH / 16) * 32;     // 57344
        k_pack_b<<<(t1 + 255) / 256, 256, 0, stream>>>(W1, W1pk, DD, HH, LL);
        int t2 = LL * (HH / 32) * (DD / 16) * 32;     // 57344
        k_pack_b<<<(t2 + 255) / 256, 256, 0, stream>>>(W2, W2pk, HH, DD, LL);
        int t3 = (DD / 32) * (DD / 16) * 32;          // 1024
        k_pack_b<<<(t3 + 255) / 256, 256, 0, stream>>>(lin_w, lwpk, DD, DD, 1);
    }

    // node encoder
    k_node_enc<<<(NN * DD + 255) / 256, 256, 0, stream>>>(x, node_w, node_b, h);

    const int edgeBlocks = (EE * 32 + 255) / 256;
    const int segBlocks  = (NN * DD + 255) / 256;
    const int tileBlocks = NN / 16;   // 20000 / 16 = 1250
    const int lnBlocks   = (NN + 7) / 8;

    for (int l = 0; l < LL; ++l) {
        const float* zin;
        if (l == 0) {
            zin = h;  // layer 0: conv directly on encoder output, no pre-norm
        } else {
            k_prenorm<<<lnBlocks, 256, 0, stream>>>(h, norm_g + l * DD, norm_b + l * DD, zb);
            zin = zb;
        }
        k_init_seg<<<segBlocks, 256, 0, stream>>>(smax, sumex, wsum);
        k_edge_max<<<edgeBlocks, 256, 0, stream>>>(zin, eattr, ei, edge_w, edge_b, t, l, smax);
        k_edge_expsum<<<edgeBlocks, 256, 0, stream>>>(zin, eattr, ei, edge_w, edge_b, t, l,
                                                      smax, sumex, wsum);
        k_mlp<<<tileBlocks, 256, 0, stream>>>(zin, wsum, sumex,
                                              W1pk + (size_t)l * DD * HH, b1 + l * HH,
                                              g1 + l * HH, be1 + l * HH,
                                              W2pk + (size_t)l * HH * DD, b2 + l * DD,
                                              h, l == 0 ? 0 : 1);
    }

    // final norm (layer 0's norm params) + output linear
    k_prenorm<<<lnBlocks, 256, 0, stream>>>(h, norm_g, norm_b, zb);
    k_final<<<tileBlocks, 256, 0, stream>>>(zb, lwpk, lin_b, out);
}
